// AttnGCN_40226663694884
// MI455X (gfx1250) — matrix operand order repaired and checked by reading, not compiled, not run on hardware
//
#include <hip/hip_runtime.h>
#include <hip/hip_bf16.h>

typedef __attribute__((ext_vector_type(16))) __bf16 v16bf;
typedef __attribute__((ext_vector_type(8)))  float  v8f;

#define HC 256   // H*C
#define CC 128   // channels per head

__device__ __forceinline__ unsigned short f2bf(float f) {
    unsigned u = __float_as_uint(f);
    unsigned r = u + 0x7FFFu + ((u >> 16) & 1u);   // round-to-nearest-even
    return (unsigned short)(r >> 16);
}
__device__ __forceinline__ unsigned fenc(float f) {
    unsigned u = __float_as_uint(f);
    return (u & 0x80000000u) ? ~u : (u | 0x80000000u);
}
__device__ __forceinline__ float fdec(unsigned e) {
    unsigned v = (e & 0x80000000u) ? (e & 0x7FFFFFFFu) : ~e;
    return __uint_as_float(v);
}

// ---------- edge-weight mean (sum; scaled by 1/E at use site) ----------
__global__ __launch_bounds__(256)
void k_mean_sum(const float* __restrict__ ew, float* __restrict__ meansum, int E) {
    __shared__ float sm[256];
    float v = 0.f;
    for (int i = blockIdx.x * 256 + threadIdx.x; i < E; i += gridDim.x * 256) v += ew[i];
    sm[threadIdx.x] = v; __syncthreads();
    for (int s = 128; s > 0; s >>= 1) {
        if ((int)threadIdx.x < s) sm[threadIdx.x] += sm[threadIdx.x + s];
        __syncthreads();
    }
    if (threadIdx.x == 0) unsafeAtomicAdd(meansum, sm[0]);
}

// ---------- in-degree (integer atomics: exact & order-independent) ----------
__global__ __launch_bounds__(256)
void k_deg(const int* __restrict__ ei, unsigned* __restrict__ deg, int E, int Et) {
    int e = blockIdx.x * 256 + threadIdx.x;
    if (e >= Et) return;
    int d = (e < E) ? ei[E + e] : (e - E);
    atomicAdd(&deg[d], 1u);
}

// ---------- activations f32 -> bf16 row-major (K zero-padded via memset) ----------
__global__ __launch_bounds__(256)
void k_x2bf(const float* __restrict__ src, unsigned short* __restrict__ dst,
            int N, int din, int Kp) {
    int i = blockIdx.x * 256 + threadIdx.x;
    if (i >= N * din) return;
    int r = i / din, c = i - r * din;
    dst[(size_t)r * Kp + c] = f2bf(src[i]);
}

// ---------- weights f32 -> bf16, packed directly into WMMA B-fragment order ----
// Bp index = ((kt*16 + tn)*32 + lane)*16 + e, where this code's contiguous B 32x16 layout:
//   n = lane&15, half = lane>>4, K = kt*32 + 16*half + e
__global__ __launch_bounds__(256)
void k_wpack(const float* __restrict__ W, unsigned short* __restrict__ Bp,
             int din, int KP) {
    int i = blockIdx.x * 256 + threadIdx.x;
    if (i >= KP * HC) return;
    int e    = i & 15;
    int lane = (i >> 4) & 31;
    int tn   = (i >> 9) & 15;
    int kt   = i >> 13;
    int n = lane & 15, half = lane >> 4;
    int K = kt * 32 + 16 * half + e;
    unsigned short v = 0;
    if (K < din) v = f2bf(W[K * HC + tn * 16 + n]);
    Bp[i] = v;
}

// ---------- WMMA GEMM: [N x KP](bf16) @ packed[KP x 256](bf16) + bias -> f32 ----
union Frag { uint4 u2[2]; v16bf v; };

template<int KP>
__global__ __launch_bounds__(128)
void k_gemm_bf16(const unsigned short* __restrict__ A,
                 const unsigned short* __restrict__ Bp,
                 const float* __restrict__ bias,
                 float* __restrict__ C,
                 int N, int rowTiles) {
    int wave = (blockIdx.x * 128 + threadIdx.x) >> 5;
    int lane = threadIdx.x & 31;
    if (wave >= rowTiles * 4) return;
    int tm = wave >> 2;        // 16-row tile
    int tg = wave & 3;         // group of 4 col tiles: tn = tg*4 .. tg*4+3
    int half = lane >> 4;
    int idx  = lane & 15;

    int arow = tm * 16 + idx;
    if (arow >= N) arow = N - 1;                 // clamp (stores are guarded)
    const unsigned short* Arow = A + (size_t)arow * KP;

    v8f acc[4] = {};
    #pragma unroll
    for (int k0 = 0; k0 < KP; k0 += 32) {
        Frag a;
        // A 16x32 bf16 layout: K=k0+16*half+e (matches packed B)
        a.u2[0] = *(const uint4*)(Arow + k0 + 16 * half);
        a.u2[1] = *(const uint4*)(Arow + k0 + 16 * half + 8);
        const int kt = k0 >> 5;
        #pragma unroll
        for (int t = 0; t < 4; ++t) {
            const unsigned short* bp =
                Bp + ((size_t)((kt * 16) + (tg * 4 + t)) * 32 + lane) * 16;
            Frag b;
            b.u2[0] = *(const uint4*)(bp);
            b.u2[1] = *(const uint4*)(bp + 8);
            acc[t] = __builtin_amdgcn_wmma_f32_16x16x32_bf16(
                        false, a.v, false, b.v, (short)0, acc[t], false, false);
        }
    }

    const bool full = (tm * 16 + 15) < N;        // wave-uniform branch
    #pragma unroll
    for (int t = 0; t < 4; ++t) {
        int col = (tg * 4 + t) * 16 + idx;
        float bi = bias[col];
        if (full) {
            #pragma unroll
            for (int i = 0; i < 8; ++i)
                C[(size_t)(tm * 16 + i + 8 * half) * HC + col] = acc[t][i] + bi;
        } else {
            #pragma unroll
            for (int i = 0; i < 8; ++i) {
                int row = tm * 16 + i + 8 * half;
                if (row < N) C[(size_t)row * HC + col] = acc[t][i] + bi;
            }
        }
    }
}

// ---------- wave-per-edge: attention logits + segment max ----------
__global__ __launch_bounds__(256)
void k_edge_logits(const int* __restrict__ ei, const float* __restrict__ ew,
                   const float* __restrict__ meansum, float invE,
                   const float* __restrict__ xl, const float* __restrict__ xr,
                   const float* __restrict__ We, const float* __restrict__ att,
                   float* __restrict__ logit, unsigned* __restrict__ segmax,
                   int E, int Et) {
    int e = (blockIdx.x * 256 + threadIdx.x) >> 5;
    int lane = threadIdx.x & 31;
    if (e >= Et) return;
    int s, d; float ea;
    if (e < E) { s = ei[e]; d = ei[E + e]; ea = ew[e]; }
    else       { s = d = e - E;            ea = meansum[0] * invE; }
    const float* xls = xl + (size_t)s * HC;
    const float* xrd = xr + (size_t)d * HC;
    #pragma unroll
    for (int h = 0; h < 2; ++h) {
        float acc = 0.f;
        #pragma unroll
        for (int cc = 0; cc < 4; ++cc) {
            int c = h * CC + cc * 32 + lane;
            float z = xls[c] + xrd[c] + ea * We[c];
            z = (z > 0.f) ? z : 0.2f * z;          // leaky 0.2 (GATv2 att)
            acc += z * att[c];
        }
        #pragma unroll
        for (int off = 16; off > 0; off >>= 1) acc += __shfl_xor(acc, off, 32);
        if (lane == 0) {
            logit[(size_t)e * 2 + h] = acc;
            atomicMax(&segmax[(size_t)d * 2 + h], fenc(acc));
        }
    }
}

// ---------- exp(logit - max) + segment sum (p overwrites logit) ----------
__global__ __launch_bounds__(256)
void k_edge_p(const int* __restrict__ ei, float* __restrict__ logit,
              const unsigned* __restrict__ segmax, float* __restrict__ segsum,
              int E, int Et) {
    int t = blockIdx.x * 256 + threadIdx.x;
    if (t >= Et * 2) return;
    int e = t >> 1, h = t & 1;
    int d = (e < E) ? ei[E + e] : (e - E);
    float p = __expf(logit[t] - fdec(segmax[(size_t)d * 2 + h]));
    logit[t] = p;
    unsafeAtomicAdd(&segsum[(size_t)d * 2 + h], p);
}

// ---------- wave-per-edge: alpha-weighted scatter-add of xl[src] ----------
__global__ __launch_bounds__(256)
void k_edge_agg(const int* __restrict__ ei, const float* __restrict__ p,
                const float* __restrict__ segsum, const float* __restrict__ xl,
                float* __restrict__ agg, int E, int Et) {
    int e = (blockIdx.x * 256 + threadIdx.x) >> 5;
    int lane = threadIdx.x & 31;
    if (e >= Et) return;
    int s = (e < E) ? ei[e]     : (e - E);
    int d = (e < E) ? ei[E + e] : (e - E);
    const float* xls = xl + (size_t)s * HC;
    float* ag = agg + (size_t)d * HC;
    #pragma unroll
    for (int h = 0; h < 2; ++h) {
        float alpha = p[(size_t)e * 2 + h] / segsum[(size_t)d * 2 + h];
        #pragma unroll
        for (int cc = 0; cc < 4; ++cc) {
            int c = h * CC + cc * 32 + lane;
            unsafeAtomicAdd(&ag[c], xls[c] * alpha);
        }
    }
}

// ---------- node output: mean-aggr over deg, mean over heads, + bias ----------
__global__ __launch_bounds__(256)
void k_node_out(const float* __restrict__ agg, const unsigned* __restrict__ deg,
                const float* __restrict__ b, float* __restrict__ xn, int N) {
    int t = blockIdx.x * 256 + threadIdx.x;
    if (t >= N * CC) return;
    int n = t >> 7, c = t & 127;
    float dv = (float)deg[n];
    float v = (agg[(size_t)n * HC + c] + agg[(size_t)n * HC + CC + c]) * (0.5f / dv) + b[c];
    xn[t] = v;
}

// ---------- batchnorm statistics (per-channel sum / sumsq) ----------
__global__ __launch_bounds__(128)
void k_bn_stats(const float* __restrict__ x, float* __restrict__ stats,
                int N, int rowsPerBlock) {
    int c = threadIdx.x;
    int r0 = blockIdx.x * rowsPerBlock;
    int r1 = r0 + rowsPerBlock; if (r1 > N) r1 = N;
    float s = 0.f, ss = 0.f;
    for (int r = r0; r < r1; ++r) {
        float v = x[(size_t)r * CC + c];
        s += v; ss += v * v;
    }
    unsafeAtomicAdd(&stats[c], s);
    unsafeAtomicAdd(&stats[CC + c], ss);
}

// ---------- batchnorm apply + leaky_relu(0.01), in place ----------
__global__ __launch_bounds__(256)
void k_bn_apply(float* __restrict__ x, const float* __restrict__ stats,
                const float* __restrict__ g, const float* __restrict__ bt, int N) {
    int t = blockIdx.x * 256 + threadIdx.x;
    if (t >= N * CC) return;
    int c = t & 127;
    float invN = 1.f / (float)N;
    float mean = stats[c] * invN;
    float var  = stats[CC + c] * invN - mean * mean;
    float y = (x[t] - mean) * rsqrtf(var + 1e-5f) * g[c] + bt[c];
    x[t] = (y > 0.f) ? y : 0.01f * y;
}

// ---------- final FC [N,128] @ [128,5] + bias ----------
__global__ __launch_bounds__(256)
void k_fc(const float* __restrict__ x, const float* __restrict__ W,
          const float* __restrict__ bias, float* __restrict__ out, int N) {
    int t = blockIdx.x * 256 + threadIdx.x;
    if (t >= N * 5) return;
    int n = t / 5, o = t - n * 5;
    const float* xrow = x + (size_t)n * CC;
    float acc = bias[o];
    #pragma unroll 4
    for (int k = 0; k < CC; ++k) acc += xrow[k] * W[k * 5 + o];
    out[t] = acc;
}

extern "C" void kernel_launch(void* const* d_in, const int* in_sizes, int n_in,
                              void* d_out, int out_size, void* d_ws, size_t ws_size,
                              hipStream_t stream) {
    const float* h  = (const float*)d_in[0];
    const int*   ei = (const int*)d_in[1];
    const float* ew = (const float*)d_in[2];
    const int N  = in_sizes[0] / 6;
    const int E  = in_sizes[2];
    const int Et = E + N;

    const float *Wl[3], *bl[3], *Wr[3], *br[3], *We[3], *att[3], *bb[3], *g[3], *bt[3];
    for (int l = 0; l < 3; ++l) {
        int base = 3 + l * 9;
        Wl[l]  = (const float*)d_in[base + 0];
        bl[l]  = (const float*)d_in[base + 1];
        Wr[l]  = (const float*)d_in[base + 2];
        br[l]  = (const float*)d_in[base + 3];
        We[l]  = (const float*)d_in[base + 4];
        att[l] = (const float*)d_in[base + 5];
        bb[l]  = (const float*)d_in[base + 6];
        g[l]   = (const float*)d_in[base + 7];
        bt[l]  = (const float*)d_in[base + 8];
    }
    const float* fcW = (const float*)d_in[30];
    const float* fcb = (const float*)d_in[31];

    // -------- workspace carve-out (256B aligned slices) --------
    char* wp = (char*)d_ws;
    auto alloc = [&](size_t bytes) -> void* {
        void* p = (void*)wp;
        wp += (bytes + 255) & ~(size_t)255;
        return p;
    };
    float*          meansum = (float*)alloc(4);
    unsigned*       deg     = (unsigned*)alloc((size_t)N * 4);
    unsigned*       segmax  = (unsigned*)alloc((size_t)N * 2 * 4);
    float*          segsum  = (float*)alloc((size_t)N * 2 * 4);
    float*          logit   = (float*)alloc((size_t)Et * 2 * 4);
    float*          xl      = (float*)alloc((size_t)N * HC * 4);
    float*          xr      = (float*)alloc((size_t)N * HC * 4);
    float*          agg     = (float*)alloc((size_t)N * HC * 4);
    float*          xA      = (float*)alloc((size_t)N * CC * 4);
    float*          xB      = (float*)alloc((size_t)N * CC * 4);
    unsigned short* xbf     = (unsigned short*)alloc((size_t)N * CC * 2);
    unsigned short* wpackL  = (unsigned short*)alloc((size_t)CC * HC * 2);
    unsigned short* wpackR  = (unsigned short*)alloc((size_t)CC * HC * 2);
    float*          stats   = (float*)alloc(HC * 4);

    // -------- per-call init --------
    hipMemsetAsync(meansum, 0, 4, stream);
    hipMemsetAsync(deg,  0, (size_t)N * 4, stream);
    hipMemsetAsync(xbf,  0, (size_t)N * CC * 2, stream);   // zero-pad K for layer 0

    k_mean_sum<<<256, 256, 0, stream>>>(ew, meansum, E);
    k_deg<<<(Et + 255) / 256, 256, 0, stream>>>(ei, deg, E, Et);

    float* xout[3] = { xA, xB, xA };
    for (int l = 0; l < 3; ++l) {
        const int din = (l == 0) ? 6 : CC;
        const int Kp  = (l == 0) ? 32 : CC;
        const float* src = (l == 0) ? h : xout[l - 1];

        hipMemsetAsync(segmax, 0, (size_t)N * 2 * 4, stream);
        hipMemsetAsync(segsum, 0, (size_t)N * 2 * 4, stream);
        hipMemsetAsync(agg,    0, (size_t)N * HC * 4, stream);
        hipMemsetAsync(stats,  0, HC * 4, stream);

        k_x2bf<<<(N * din + 255) / 256, 256, 0, stream>>>(src, xbf, N, din, Kp);
        k_wpack<<<(Kp * HC + 255) / 256, 256, 0, stream>>>(Wl[l], wpackL, din, Kp);
        k_wpack<<<(Kp * HC + 255) / 256, 256, 0, stream>>>(Wr[l], wpackR, din, Kp);

        const int rowTiles = (N + 15) / 16;
        const int gemmBlocks = (rowTiles * 4 * 32 + 127) / 128;  // 4 waves/blk, 4 col-tiles/wave
        if (l == 0) {
            k_gemm_bf16<32><<<gemmBlocks, 128, 0, stream>>>(xbf, wpackL, bl[l], xl, N, rowTiles);
            k_gemm_bf16<32><<<gemmBlocks, 128, 0, stream>>>(xbf, wpackR, br[l], xr, N, rowTiles);
        } else {
            k_gemm_bf16<128><<<gemmBlocks, 128, 0, stream>>>(xbf, wpackL, bl[l], xl, N, rowTiles);
            k_gemm_bf16<128><<<gemmBlocks, 128, 0, stream>>>(xbf, wpackR, br[l], xr, N, rowTiles);
        }

        const int eBlocks = (int)(((long long)Et * 32 + 255) / 256);
        k_edge_logits<<<eBlocks, 256, 0, stream>>>(ei, ew, meansum, 1.0f / (float)E,
                                                   xl, xr, We[l], att[l],
                                                   logit, segmax, E, Et);
        k_edge_p<<<(Et * 2 + 255) / 256, 256, 0, stream>>>(ei, logit, segmax, segsum, E, Et);
        k_edge_agg<<<eBlocks, 256, 0, stream>>>(ei, logit, segsum, xl, agg, E, Et);

        k_node_out<<<(N * CC + 255) / 256, 256, 0, stream>>>(agg, deg, bb[l], xout[l], N);

        const int bnBlocks = 120;
        k_bn_stats<<<bnBlocks, 128, 0, stream>>>(xout[l], stats, N, (N + bnBlocks - 1) / bnBlocks);
        k_bn_apply<<<(N * CC + 255) / 256, 256, 0, stream>>>(xout[l], stats, g[l], bt[l], N);
    }

    k_fc<<<(N * 5 + 255) / 256, 256, 0, stream>>>(xout[2], fcW, fcb, (float*)d_out, N);
}
